// SSSDS4Imputer_87376814669950
// MI455X (gfx1250) — matrix operand order repaired and checked by reading, not compiled, not run on hardware
//
#include <hip/hip_runtime.h>

// ---------------------------------------------------------------------------
// SSSD-S4 imputer forward for MI455X (gfx1250, wave32, WMMA bf16 + TDM).
//
// Activations live as (C, N) row-major, N = B*L = 16384.  All GEMM operands
// (weights + activation panels) are stored bf16 in the workspace; GEMM tiles
// are staged into LDS by the Tensor Data Mover (TENSOR_LOAD_TO_LDS with LDS
// padding for the A tile), synchronized with s_wait_tensorcnt.  The S4 long
// convolution is an exact diagonal-SSM bidirectional recurrence scan.
// ---------------------------------------------------------------------------

#define L_SEQ 1024
#define BATCH 16
#define NTOT  (BATCH * L_SEQ)   // 16384 = 1<<14
#define RESC  256
#define HF    512
#define NHM   32
#define INC   14
#define NLAY  10

typedef __attribute__((ext_vector_type(16))) __bf16 bf16x16;
typedef __attribute__((ext_vector_type(8)))  float  v8f;

static __device__ __forceinline__ float sigf(float x)  { return 1.f / (1.f + __expf(-x)); }
static __device__ __forceinline__ float siluf(float x) { return x * sigf(x); }
static __device__ __forceinline__ float geluf(float x) { return 0.5f * x * (1.f + erff(x * 0.70710678118f)); }

static __device__ __forceinline__ unsigned short f2bf(float f) {
  unsigned int u = __float_as_uint(f);
  unsigned int r = u + 0x7fffu + ((u >> 16) & 1u);   // round-to-nearest-even
  return (unsigned short)(r >> 16);
}

// ---------------------------------------------------------------------------
// Tensor Data Mover (TDM) 2D tile load: global (bf16) -> LDS, optional LDS
// row padding.  Descriptor layout per CDNA5 ISA section 8 (D# groups 0/1).
// ---------------------------------------------------------------------------
#if __has_builtin(__builtin_amdgcn_tensor_load_to_lds) && __has_builtin(__builtin_amdgcn_s_wait_tensorcnt)
#define HAVE_TDM 1
typedef __attribute__((ext_vector_type(4))) unsigned int u32x4;
typedef __attribute__((ext_vector_type(8))) int i32x8;
typedef __attribute__((ext_vector_type(4))) int i32x4;

static __device__ __forceinline__ void tdm_load_2d(
    unsigned lds_off, const void* gptr,
    unsigned tensor_d0, unsigned tensor_d1,          // remaining extents (elems)
    unsigned tile_w, unsigned tile_h,                // tile dims (elems)
    unsigned stride_elems,                           // row stride (elems)
    unsigned pad_interval_code, unsigned pad_amount_code, unsigned pad_en) {
  unsigned long long ga = (unsigned long long)(size_t)gptr;
  u32x4 g0;
  g0[0] = 1u;                                               // count=1, user D#
  g0[1] = lds_off;                                          // LDS byte address
  g0[2] = (unsigned)ga;                                     // global_addr[31:0]
  g0[3] = (unsigned)((ga >> 32) & 0x01ffffffu) | (2u << 30);// addr[56:32] | type=2
  i32x8 g1;
  unsigned w0 = (1u << 16);                                 // data_size = 2 bytes
  if (pad_en) w0 |= (1u << 20) | (pad_interval_code << 22) | (pad_amount_code << 25);
  g1[0] = (int)w0;
  g1[1] = (int)((tensor_d0 & 0xffffu) << 16);               // dim0 low (bits 63:48)
  g1[2] = (int)(((tensor_d0 >> 16) & 0xffffu) | ((tensor_d1 & 0xffffu) << 16));
  g1[3] = (int)(((tensor_d1 >> 16) & 0xffffu) | ((tile_w & 0xffffu) << 16));
  g1[4] = (int)(tile_h & 0xffffu);                          // tile_dim1; tile_dim2=0
  g1[5] = (int)stride_elems;                                // dim0_stride[31:0]
  g1[6] = 0;
  g1[7] = 0;
  i32x4 z4 = {0, 0, 0, 0};
#if __clang_major__ >= 23
  i32x8 z8 = {0, 0, 0, 0, 0, 0, 0, 0};
  __builtin_amdgcn_tensor_load_to_lds(g0, g1, z4, z4, z8, 0);
#else
  __builtin_amdgcn_tensor_load_to_lds(g0, g1, z4, z4, 0);
#endif
}
#else
#define HAVE_TDM 0
#endif

// ===========================================================================
// bf16 WMMA GEMM: Y(M,N) = epilogue( A(M,K) @ X(K,N) + bias ),  A/X bf16.
//   EP=0: none      EP=1: relu      EP=2: Y = (res + v) * oscale
// M%128==0, N%128==0, K%32==0 at all call sites.
// Block = 256 threads (8 waves), tile 128x128x32, waves 2(M) x 4(N),
// 8 WMMA 16x16 tiles per wave per K-step.
// ===========================================================================
#define BM 128
#define BN 128
#define BKT 32
#define PITCH 40     // halves per fragment-layout LDS row (80 B)
#define BRPITCH 136  // halves per raw B row (128 + 8 pad)

union FragBF { bf16x16 v; uint4 q[2]; };

template <int EP>
__global__ __launch_bounds__(256) void k_gemm(
    const unsigned short* __restrict__ A, const unsigned short* __restrict__ X,
    const float* __restrict__ bias, const float* __restrict__ res,
    float* __restrict__ Y, int M, int N, int K, float oscale) {
  __shared__ __align__(16) unsigned short As[BM * PITCH];      // [m][k], pitch 40
  __shared__ __align__(16) unsigned short Braw[BKT * BRPITCH]; // [k][n], pitch 136
  __shared__ __align__(16) unsigned short Bs[BN * PITCH];      // [n][k], pitch 40

  const int t    = threadIdx.x;
  const int lane = t & 31;
  const int wid  = t >> 5;
  const int wm   = wid >> 2;   // 0..1
  const int wn   = wid & 3;    // 0..3
  const int m0   = blockIdx.y * BM;
  const int n0   = blockIdx.x * BN;

  const v8f vzero = {0.f, 0.f, 0.f, 0.f, 0.f, 0.f, 0.f, 0.f};
  v8f acc[4][2];
#pragma unroll
  for (int i = 0; i < 4; ++i)
#pragma unroll
    for (int j = 0; j < 2; ++j) acc[i][j] = vzero;

  for (int kt = 0; kt < K; kt += BKT) {
    __syncthreads();   // previous K-step fully consumed before restaging
#if HAVE_TDM
    if (wid == 0) {
      // A tile: 128 rows x 32 halves, padded in LDS to pitch 40 halves
      // (pad_interval code 3 = 16 DWORDs, pad_amount code 3 = 4 DWORDs).
      tdm_load_2d((unsigned)(size_t)(void*)As, A + (size_t)m0 * K + kt,
                  (unsigned)(K - kt), (unsigned)(M - m0), 32u, 128u,
                  (unsigned)K, 3u, 3u, 1u);
      // B tile: 32 rows x 128 halves, padded to pitch 136 halves
      // (pad_interval code 5 = 64 DWORDs, pad_amount code 3 = 4 DWORDs).
      tdm_load_2d((unsigned)(size_t)(void*)Braw, X + (size_t)kt * N + n0,
                  (unsigned)(N - n0), (unsigned)(K - kt), 128u, 32u,
                  (unsigned)N, 5u, 3u, 1u);
      __builtin_amdgcn_s_wait_tensorcnt(0);
    }
    if (t == 0 && kt + BKT < K)
      __builtin_prefetch(X + (size_t)(kt + BKT) * N + n0, 0, 1);
#else
    // fallback: VGPR staging of bf16 tiles
#pragma unroll
    for (int p = 0; p < 2; ++p) {
      int r  = (t >> 2) + 64 * p;
      int c8 = (t & 3) * 8;
      uint4 q = *(const uint4*)(A + (size_t)(m0 + r) * K + kt + c8);
      *(uint4*)(&As[r * PITCH + c8]) = q;
    }
#pragma unroll
    for (int p = 0; p < 2; ++p) {
      int k  = (t >> 4) + 16 * p;
      int n8 = (t & 15) * 8;
      uint4 q = *(const uint4*)(X + (size_t)(kt + k) * N + n0 + n8);
      *(uint4*)(&Braw[k * BRPITCH + n8]) = q;
    }
#endif
    __syncthreads();
    // transpose Braw [k][n] -> Bs [n][k] (fragment layout)
    {
      int k  = t >> 3;            // 0..31
      int ng = (t & 7) * 16;      // 0..112
      union { uint4 q[2]; unsigned short h[16]; } u;
      u.q[0] = *(const uint4*)(&Braw[k * BRPITCH + ng]);
      u.q[1] = *(const uint4*)(&Braw[k * BRPITCH + ng + 8]);
#pragma unroll
      for (int e = 0; e < 16; ++e) Bs[(ng + e) * PITCH + k] = u.h[e];
    }
    __syncthreads();

    const int kbase = (lane >> 4) * 16;   // A frag: K runs kbase..kbase+15
    const int kb2   = (lane >> 4) * 16;   // B frag: K runs kb2..kb2+15
    FragBF fb[2];
#pragma unroll
    for (int j = 0; j < 2; ++j) {
      int nrow = wn * 32 + j * 16 + (lane & 15);
      const uint4* pb = (const uint4*)(&Bs[nrow * PITCH + kb2]);
      fb[j].q[0] = pb[0];
      fb[j].q[1] = pb[1];
    }
#pragma unroll
    for (int i = 0; i < 4; ++i) {
      int mrow = wm * 64 + i * 16 + (lane & 15);
      FragBF fa;
      fa.q[0] = *(const uint4*)(&As[mrow * PITCH + kbase]);
      fa.q[1] = *(const uint4*)(&As[mrow * PITCH + kbase + 8]);
#pragma unroll
      for (int j = 0; j < 2; ++j)
        acc[i][j] = __builtin_amdgcn_wmma_f32_16x16x32_bf16(
            false, fa.v, false, fb[j].v, (short)0, acc[i][j], false, false);
    }
  }

  // epilogue: D element r of lane -> (m = base + r + 8*(lane>>4), n = lane&15)
#pragma unroll
  for (int i = 0; i < 4; ++i) {
#pragma unroll
    for (int j = 0; j < 2; ++j) {
      int mb = m0 + wm * 64 + i * 16 + (lane >> 4) * 8;
      int nc = n0 + wn * 32 + j * 16 + (lane & 15);
#pragma unroll
      for (int r = 0; r < 8; ++r) {
        int m = mb + r;
        float v = acc[i][j][r] + bias[m];
        if (EP == 1) v = fmaxf(v, 0.f);
        if (EP == 2) v = (v + res[(size_t)m * N + nc]) * oscale;
        Y[(size_t)m * N + nc] = v;
      }
    }
  }
}

// ===========================================================================
// Small / elementwise kernels
// ===========================================================================

__global__ void k_zero(float* __restrict__ p, int n) {
  int i = blockIdx.x * blockDim.x + threadIdx.x;
  if (i < n) p[i] = 0.f;
}

// f32 -> bf16 converters (weights once per launch; activations per use)
__global__ void k_cvt(const float* __restrict__ s, unsigned short* __restrict__ d, int n) {
  int i = blockIdx.x * blockDim.x + threadIdx.x;
  if (i < n) d[i] = f2bf(s[i]);
}
__global__ void k_cvt_scale(const float* __restrict__ s, unsigned short* __restrict__ d,
                            float sc, int n) {
  int i = blockIdx.x * blockDim.x + threadIdx.x;
  if (i < n) d[i] = f2bf(s[i] * sc);
}
__global__ void k_gelu_cvt(const float* __restrict__ s, unsigned short* __restrict__ d, int n) {
  int i = blockIdx.x * blockDim.x + threadIdx.x;
  if (i < n) d[i] = f2bf(geluf(s[i]));
}

// cond MLP: ce = label*ceW + ceB ; c2 = gelu(ce@W1^T+b1)@W2^T+b2   (one block)
__global__ void k_cond_mlp(const float* __restrict__ label,
                           const float* __restrict__ ceW, const float* __restrict__ ceB,
                           const float* __restrict__ W1, const float* __restrict__ b1,
                           const float* __restrict__ W2, const float* __restrict__ b2,
                           float* __restrict__ c2out) {
  __shared__ float ce[BATCH * INC];
  __shared__ float c1[BATCH * 56];
  int t = threadIdx.x;
  if (t < BATCH * INC) {
    int b = t / INC, c = t % INC;
    ce[t] = label[b] * ceW[c] + ceB[c];
  }
  __syncthreads();
  for (int i = t; i < BATCH * 56; i += blockDim.x) {
    int b = i / 56, j = i % 56;
    float a = b1[j];
    for (int k = 0; k < INC; ++k) a += W1[j * INC + k] * ce[b * INC + k];
    c1[i] = geluf(a);
  }
  __syncthreads();
  if (t < BATCH * INC) {
    int b = t / INC, c = t % INC;
    float a = b2[c];
    for (int k = 0; k < 56; ++k) a += W2[c * 56 + k] * c1[b * 56 + k];
    c2out[t] = a;
  }
}

__global__ void k_cond_fill(const float* __restrict__ c2, const float* __restrict__ seqW,
                            const float* __restrict__ seqB, float* __restrict__ cond) {
  int idx = blockIdx.x * blockDim.x + threadIdx.x;
  if (idx >= INC * NTOT) return;
  int c = idx >> 14, n = idx & (NTOT - 1);
  int b = n >> 10, l = n & (L_SEQ - 1);
  cond[idx] = c2[b * INC + c] * seqW[l] + seqB[l];
}

// e1 = silu(emb_table[steps] @ p1_W^T + b)
__global__ void k_emb1(const int* __restrict__ steps, const float* __restrict__ emb,
                       const float* __restrict__ W, const float* __restrict__ bias,
                       float* __restrict__ Y) {
  int idx = blockIdx.x * blockDim.x + threadIdx.x;
  if (idx >= BATCH * 512) return;
  int b = idx / 512, m = idx % 512;
  const float* x = emb + (size_t)steps[b] * 128;
  float a = bias[m];
  for (int k = 0; k < 128; ++k) a += x[k] * W[m * 128 + k];
  Y[idx] = siluf(a);
}

// Y[b,m] = act(X[b,:]@W[m,:] + bias[m]); ACT: 0=none 1=silu
template <int ACT>
__global__ void k_dense(const float* __restrict__ X, const float* __restrict__ W,
                        const float* __restrict__ bias, float* __restrict__ Y,
                        int Bn, int M, int K) {
  int idx = blockIdx.x * blockDim.x + threadIdx.x;
  if (idx >= Bn * M) return;
  int b = idx / M, m = idx % M;
  float a = bias[m];
  const float* x = X + (size_t)b * K;
  const float* w = W + (size_t)m * K;
  for (int k = 0; k < K; ++k) a += x[k] * w[k];
  if (ACT == 1) a = siluf(a);
  Y[idx] = a;
}

// x[m,n] = relu( init_W(256x14) @ input(B,14,L) + b )   -> canonical (C,N)
__global__ void k_init(const float* __restrict__ in, const float* __restrict__ W,
                       const float* __restrict__ bias, float* __restrict__ X) {
  int idx = blockIdx.x * blockDim.x + threadIdx.x;
  if (idx >= RESC * NTOT) return;
  int m = idx >> 14, n = idx & (NTOT - 1);
  int b = n >> 10, l = n & (L_SEQ - 1);
  float a = bias[m];
  for (int k = 0; k < INC; ++k) a += W[m * INC + k] * in[(size_t)(b * INC + k) * L_SEQ + l];
  X[idx] = fmaxf(a, 0.f);
}

// im2col for k=3 conv (bf16 out), part_t broadcast added, zero padding
__global__ void k_im2col(const float* __restrict__ X, const float* __restrict__ pt,
                         unsigned short* __restrict__ XI) {
  int idx = blockIdx.x * blockDim.x + threadIdx.x;
  if (idx >= 3 * RESC * NTOT) return;
  int k = idx >> 14, n = idx & (NTOT - 1);
  int c = k / 3, tt = k - 3 * c;
  int b = n >> 10, l = n & (L_SEQ - 1);
  int ll = l + tt - 1;
  float v = 0.f;
  if (ll >= 0 && ll < L_SEQ)
    v = X[(size_t)c * NTOT + b * L_SEQ + ll] + pt[b * RESC + c];
  XI[idx] = f2bf(v);
}

// u[h,n] += cond_W(Hx14) @ cond(14,N) + cond_b
__global__ void k_condadd(const float* __restrict__ cond, const float* __restrict__ W,
                          const float* __restrict__ bias, float* __restrict__ U) {
  int idx = blockIdx.x * blockDim.x + threadIdx.x;
  if (idx >= HF * NTOT) return;
  int h = idx >> 14, n = idx & (NTOT - 1);
  float a = bias[h];
  for (int k = 0; k < INC; ++k) a += W[h * INC + k] * cond[(size_t)k * NTOT + n];
  U[idx] += a;
}

// out = bf16( tanh(u[:256]) * sigmoid(u[256:]) )
__global__ void k_gate(const float* __restrict__ U, unsigned short* __restrict__ O) {
  int idx = blockIdx.x * blockDim.x + threadIdx.x;
  if (idx >= RESC * NTOT) return;
  int c = idx >> 14, n = idx & (NTOT - 1);
  O[idx] = f2bf(tanhf(U[(size_t)c * NTOT + n]) * sigf(U[(size_t)(c + RESC) * NTOT + n]));
}

// S4 coefficients: w = exp(dt*A), cb_c = 2 * C_c * B*(w-1)/A   per (h,n)
__global__ void k_s4_coef(const float* __restrict__ log_dt,
                          const float* __restrict__ Are, const float* __restrict__ Aim,
                          const float* __restrict__ Bre, const float* __restrict__ Bim,
                          const float* __restrict__ Cre, const float* __restrict__ Cim,
                          float* __restrict__ coef) {
  int idx = blockIdx.x * blockDim.x + threadIdx.x;
  if (idx >= HF * NHM) return;
  int h = idx >> 5;
  float dt = __expf(log_dt[h]);
  float ar = Are[idx], ai = Aim[idx];
  float xr = ar * dt, xi = ai * dt;
  float er = __expf(xr);
  float wre = er * __cosf(xi), wim = er * __sinf(xi);
  float nr = wre - 1.f, ni = wim;
  float den = ar * ar + ai * ai;
  float ir = (nr * ar + ni * ai) / den;
  float ii = (ni * ar - nr * ai) / den;
  float br = Bre[idx], bi = Bim[idx];
  float dbr = br * ir - bi * ii;
  float dbi = br * ii + bi * ir;
  const int S = HF * NHM;
  coef[idx]         = wre;
  coef[S + idx]     = wim;
  float c0r = Cre[idx], c0i = Cim[idx];
  coef[2 * S + idx] = 2.f * (c0r * dbr - c0i * dbi);
  coef[3 * S + idx] = 2.f * (c0r * dbi + c0i * dbr);
  float c1r = Cre[S + idx], c1i = Cim[S + idx];
  coef[4 * S + idx] = 2.f * (c1r * dbr - c1i * dbi);
  coef[5 * S + idx] = 2.f * (c1r * dbi + c1i * dbr);
}

// Bidirectional diagonal-SSM scan.  One thread per (b,h); 32 complex modes
// in registers; streams a contiguous 4KB row of u and y.
__global__ __launch_bounds__(64) void k_s4_scan(const float* __restrict__ u,
                                                float* __restrict__ y,
                                                const float* __restrict__ coef,
                                                const float* __restrict__ Dp) {
  int idx = blockIdx.x * blockDim.x + threadIdx.x;   // 0..8191
  int b = idx & 15;
  int h = idx >> 4;
  const int S = HF * NHM;
  const int cb = h * NHM;
  float wr[NHM], wi[NHM], cr[NHM], ci[NHM], sr[NHM], si[NHM];
#pragma unroll
  for (int n = 0; n < NHM; ++n) {
    wr[n] = coef[cb + n];       wi[n] = coef[S + cb + n];
    cr[n] = coef[2*S + cb + n]; ci[n] = coef[3*S + cb + n];
    sr[n] = 0.f; si[n] = 0.f;
  }
  const float dD = Dp[h];
  const float* up = u + (size_t)h * NTOT + b * L_SEQ;
  float*       yp = y + (size_t)h * NTOT + b * L_SEQ;
  for (int l = 0; l < L_SEQ; ++l) {
    float uv = up[l];
    float acc = dD * uv;
#pragma unroll
    for (int n = 0; n < NHM; ++n) {
      float nr2 = wr[n] * sr[n] - wi[n] * si[n] + uv;
      float ni2 = wr[n] * si[n] + wi[n] * sr[n];
      sr[n] = nr2; si[n] = ni2;
      acc += cr[n] * nr2 - ci[n] * ni2;
    }
    yp[l] = acc;
  }
#pragma unroll
  for (int n = 0; n < NHM; ++n) {
    cr[n] = coef[4*S + cb + n]; ci[n] = coef[5*S + cb + n];
    sr[n] = 0.f; si[n] = 0.f;
  }
  for (int l = L_SEQ - 2; l >= 0; --l) {
    float uv = up[l + 1];
    float acc = 0.f;
#pragma unroll
    for (int n = 0; n < NHM; ++n) {
      float nr2 = wr[n] * sr[n] - wi[n] * si[n] + uv;
      float ni2 = wr[n] * si[n] + wi[n] * sr[n];
      sr[n] = nr2; si[n] = ni2;
      acc += cr[n] * nr2 - ci[n] * ni2;
    }
    yp[l] += acc;
  }
}

// GLU(a,g) + residual, then LayerNorm over the H channel dim, in place on U.
__global__ __launch_bounds__(256) void k_glu_ln(const float* __restrict__ T,
                                                float* __restrict__ U,
                                                const float* __restrict__ lng,
                                                const float* __restrict__ lnb) {
  __shared__ float ssum[4][64], ssq[4][64], smu[64], sri[64];
  int tx = threadIdx.x, ty = threadIdx.y;
  int n = blockIdx.x * 64 + tx;
  float sum = 0.f, sq = 0.f;
  for (int h = ty; h < HF; h += 4) {
    float a = T[(size_t)h * NTOT + n];
    float g = T[(size_t)(h + HF) * NTOT + n];
    float v = a * sigf(g) + U[(size_t)h * NTOT + n];
    U[(size_t)h * NTOT + n] = v;
    sum += v; sq += v * v;
  }
  ssum[ty][tx] = sum; ssq[ty][tx] = sq;
  __syncthreads();
  if (ty == 0) {
    float s = ssum[0][tx] + ssum[1][tx] + ssum[2][tx] + ssum[3][tx];
    float q = ssq[0][tx] + ssq[1][tx] + ssq[2][tx] + ssq[3][tx];
    float mu = s * (1.f / HF);
    float var = q * (1.f / HF) - mu * mu;
    smu[tx] = mu; sri[tx] = rsqrtf(var + 1e-5f);
  }
  __syncthreads();
  float mu = smu[tx], ri = sri[tx];
  for (int h = ty; h < HF; h += 4) {
    float v = U[(size_t)h * NTOT + n];
    U[(size_t)h * NTOT + n] = (v - mu) * ri * lng[h] + lnb[h];
  }
}

// final projection z_W(14x256) and transform back to (B, OUT_C, L) layout
__global__ void k_zfinal(const float* __restrict__ Yin, const float* __restrict__ W,
                         const float* __restrict__ bias, float* __restrict__ out) {
  int idx = blockIdx.x * blockDim.x + threadIdx.x;
  if (idx >= INC * NTOT) return;
  int c = idx >> 14, n = idx & (NTOT - 1);
  int b = n >> 10, l = n & (L_SEQ - 1);
  float a = bias[c];
  for (int k = 0; k < RESC; ++k) a += W[c * RESC + k] * Yin[(size_t)k * NTOT + n];
  out[(size_t)(b * INC + c) * L_SEQ + l] = a;
}

// ===========================================================================
// Host side
// ===========================================================================
static void run_s4(hipStream_t stream, float* u, float* y, float* tbuf, float* coef,
                   unsigned short* xbf, const unsigned short* outWbf,
                   const float* log_dt, const float* Are, const float* Aim,
                   const float* Bre, const float* Bim,
                   const float* Cre, const float* Cim, const float* Dp,
                   const float* outB, const float* lng, const float* lnb) {
  k_s4_coef<<<(HF * NHM + 255) / 256, 256, 0, stream>>>(log_dt, Are, Aim, Bre, Bim, Cre, Cim, coef);
  k_s4_scan<<<(BATCH * HF) / 64, 64, 0, stream>>>(u, y, coef, Dp);
  k_gelu_cvt<<<(HF * NTOT + 255) / 256, 256, 0, stream>>>(y, xbf, HF * NTOT);
  k_gemm<0><<<dim3(NTOT / 128, (2 * HF) / 128), 256, 0, stream>>>(
      outWbf, xbf, outB, nullptr, tbuf, 2 * HF, NTOT, HF, 1.f);
  k_glu_ln<<<dim3(NTOT / 64), dim3(64, 4), 0, stream>>>(tbuf, u, lng, lnb);
}

extern "C" void kernel_launch(void* const* d_in, const int* in_sizes, int n_in,
                              void* d_out, int out_size, void* d_ws, size_t ws_size,
                              hipStream_t stream) {
  (void)in_sizes; (void)n_in; (void)out_size; (void)ws_size;
  // -- input table (recursive insertion order of setup_inputs()) --
  const float* input   = (const float*)d_in[0];
  const int*   steps   = (const int*)  d_in[1];
  const float* label   = (const float*)d_in[2];
  const float* ceW     = (const float*)d_in[3];
  const float* ceB     = (const float*)d_in[4];
  const float* cW1     = (const float*)d_in[6];
  const float* cb1     = (const float*)d_in[7];
  const float* cW2     = (const float*)d_in[8];
  const float* cb2     = (const float*)d_in[9];
  const float* seqW    = (const float*)d_in[10];
  const float* seqB    = (const float*)d_in[11];
  const float* initW   = (const float*)d_in[12];
  const float* initB   = (const float*)d_in[13];
  const float* p1W     = (const float*)d_in[14];
  const float* p1B     = (const float*)d_in[15];
  const float* p2W     = (const float*)d_in[16];
  const float* p2B     = (const float*)d_in[17];
  const float* fctW    = (const float*)d_in[18];
  const float* fctB    = (const float*)d_in[19];
  const float* cvW     = (const float*)d_in[20];
  const float* cvB     = (const float*)d_in[21];
  const float* cndW    = (const float*)d_in[22];
  const float* cndB    = (const float*)d_in[23];
  const float* resW    = (const float*)d_in[24];
  const float* resB    = (const float*)d_in[25];
  const float* skpW    = (const float*)d_in[26];
  const float* skpB    = (const float*)d_in[27];
  const float* s4p[2][12];
  for (int s = 0; s < 2; ++s)
    for (int k = 0; k < 12; ++k) s4p[s][k] = (const float*)d_in[28 + s * 12 + k];
  const float* f1W     = (const float*)d_in[52];
  const float* f1B     = (const float*)d_in[53];
  const float* zW      = (const float*)d_in[54];
  const float* zB      = (const float*)d_in[55];
  const float* embT    = (const float*)d_in[56];
  float* out = (float*)d_out;

  // -- workspace layout --
  float* ws = (float*)d_ws;
  const size_t N = NTOT;
  float* ws_x    = ws;                         // 256*N f32
  float* ws_u    = ws_x    + (size_t)RESC * N; // 512*N
  float* ws_y    = ws_u    + (size_t)HF   * N; // 512*N
  float* ws_t    = ws_y    + (size_t)HF   * N; // 1024*N (out_W result)
  float* ws_skip = ws_t    + (size_t)2*HF * N; // 256*N
  float* ws_cond = ws_skip + (size_t)RESC * N; // 14*N
  float* ws_c2   = ws_cond + (size_t)INC  * N; // 256
  float* ws_e1   = ws_c2   + 256;              // 8192
  float* ws_e    = ws_e1   + 8192;             // 8192
  float* ws_pt   = ws_e    + 8192;             // 4096
  float* ws_coef = ws_pt   + 4096;             // 6*512*32

  // bf16 arenas (16B-aligned)
  uintptr_t bp = (uintptr_t)(ws_coef + 6 * HF * NHM);
  bp = (bp + 15) & ~(uintptr_t)15;
  unsigned short* ws_bx  = (unsigned short*)bp;            // 768*N halves (activations)
  const size_t ncv = (size_t)NLAY * HF * 3 * RESC;         // 3,932,160
  const size_t now_ = (size_t)NLAY * 2 * HF * HF;          // 5,242,880
  const size_t nrs = (size_t)NLAY * RESC * RESC;           // 655,360
  unsigned short* wbf_cv  = ws_bx   + (size_t)3 * RESC * N;
  unsigned short* wbf_o1  = wbf_cv  + ncv;
  unsigned short* wbf_o2  = wbf_o1  + now_;
  unsigned short* wbf_res = wbf_o2  + now_;
  unsigned short* wbf_skp = wbf_res + nrs;
  unsigned short* wbf_f1  = wbf_skp + nrs;                 // 65536

  const int TB = 256;
  const float SQ5 = 0.70710678118f;            // sqrt(0.5)
  const float SQN = 0.31622776601f;            // sqrt(1/NL)

  // pre-convert all GEMM weights to bf16 (once per launch)
  k_cvt<<<(int)((ncv + TB - 1) / TB), TB, 0, stream>>>(cvW, wbf_cv, (int)ncv);
  k_cvt<<<(int)((now_ + TB - 1) / TB), TB, 0, stream>>>(s4p[0][8], wbf_o1, (int)now_);
  k_cvt<<<(int)((now_ + TB - 1) / TB), TB, 0, stream>>>(s4p[1][8], wbf_o2, (int)now_);
  k_cvt<<<(int)((nrs + TB - 1) / TB), TB, 0, stream>>>(resW, wbf_res, (int)nrs);
  k_cvt<<<(int)((nrs + TB - 1) / TB), TB, 0, stream>>>(skpW, wbf_skp, (int)nrs);
  k_cvt<<<(RESC * RESC + TB - 1) / TB, TB, 0, stream>>>(f1W, wbf_f1, RESC * RESC);

  // conditioning + embeddings + init
  k_cond_mlp<<<1, 256, 0, stream>>>(label, ceW, ceB, cW1, cb1, cW2, cb2, ws_c2);
  k_cond_fill<<<(INC * NTOT + TB - 1) / TB, TB, 0, stream>>>(ws_c2, seqW, seqB, ws_cond);
  k_emb1<<<(BATCH * 512 + TB - 1) / TB, TB, 0, stream>>>(steps, embT, p1W, p1B, ws_e1);
  k_dense<1><<<(BATCH * 512 + TB - 1) / TB, TB, 0, stream>>>(ws_e1, p2W, p2B, ws_e, BATCH, 512, 512);
  k_init<<<(RESC * NTOT + TB - 1) / TB, TB, 0, stream>>>(input, initW, initB, ws_x);
  k_zero<<<(RESC * NTOT + TB - 1) / TB, TB, 0, stream>>>(ws_skip, RESC * NTOT);

  for (int i = 0; i < NLAY; ++i) {
    k_dense<0><<<(BATCH * RESC + TB - 1) / TB, TB, 0, stream>>>(
        ws_e, fctW + (size_t)i * RESC * 512, fctB + i * RESC, ws_pt, BATCH, RESC, 512);
    // conv k3 as GEMM(512 x 768) over bf16 im2col
    k_im2col<<<(3 * RESC * NTOT + TB - 1) / TB, TB, 0, stream>>>(ws_x, ws_pt, ws_bx);
    k_gemm<0><<<dim3(NTOT / 128, HF / 128), 256, 0, stream>>>(
        wbf_cv + (size_t)i * HF * 3 * RESC, ws_bx, cvB + i * HF, nullptr,
        ws_u, HF, NTOT, 3 * RESC, 1.f);
    // S4 #1
    run_s4(stream, ws_u, ws_y, ws_t, ws_coef, ws_bx,
           wbf_o1 + (size_t)i * 2 * HF * HF,
           s4p[0][0] + (size_t)i * HF, s4p[0][1] + (size_t)i * HF * NHM,
           s4p[0][2] + (size_t)i * HF * NHM, s4p[0][3] + (size_t)i * HF * NHM,
           s4p[0][4] + (size_t)i * HF * NHM, s4p[0][5] + (size_t)i * 2 * HF * NHM,
           s4p[0][6] + (size_t)i * 2 * HF * NHM, s4p[0][7] + (size_t)i * HF,
           s4p[0][9] + (size_t)i * 2 * HF, s4p[0][10] + (size_t)i * HF,
           s4p[0][11] + (size_t)i * HF);
    // + cond projection
    k_condadd<<<(HF * NTOT + TB - 1) / TB, TB, 0, stream>>>(
        ws_cond, cndW + (size_t)i * HF * INC, cndB + i * HF, ws_u);
    // S4 #2
    run_s4(stream, ws_u, ws_y, ws_t, ws_coef, ws_bx,
           wbf_o2 + (size_t)i * 2 * HF * HF,
           s4p[1][0] + (size_t)i * HF, s4p[1][1] + (size_t)i * HF * NHM,
           s4p[1][2] + (size_t)i * HF * NHM, s4p[1][3] + (size_t)i * HF * NHM,
           s4p[1][4] + (size_t)i * HF * NHM, s4p[1][5] + (size_t)i * 2 * HF * NHM,
           s4p[1][6] + (size_t)i * 2 * HF * NHM, s4p[1][7] + (size_t)i * HF,
           s4p[1][9] + (size_t)i * 2 * HF, s4p[1][10] + (size_t)i * HF,
           s4p[1][11] + (size_t)i * HF);
    // gate (bf16), residual, skip
    k_gate<<<(RESC * NTOT + TB - 1) / TB, TB, 0, stream>>>(ws_u, ws_bx);
    k_gemm<2><<<dim3(NTOT / 128, RESC / 128), 256, 0, stream>>>(
        wbf_res + (size_t)i * RESC * RESC, ws_bx, resB + i * RESC, ws_x,
        ws_x, RESC, NTOT, RESC, SQ5);
    k_gemm<2><<<dim3(NTOT / 128, RESC / 128), 256, 0, stream>>>(
        wbf_skp + (size_t)i * RESC * RESC, ws_bx, skpB + i * RESC, ws_skip,
        ws_skip, RESC, NTOT, RESC, 1.f);
  }

  // final: y = relu(f1_W @ (skip*sqrt(1/NL)) + b); out = z_W @ y + z_b
  k_cvt_scale<<<(RESC * NTOT + TB - 1) / TB, TB, 0, stream>>>(ws_skip, ws_bx, SQN, RESC * NTOT);
  k_gemm<1><<<dim3(NTOT / 128, RESC / 128), 256, 0, stream>>>(
      wbf_f1, ws_bx, f1B, nullptr, ws_y, RESC, NTOT, RESC, 1.f);
  k_zfinal<<<(INC * NTOT + TB - 1) / TB, TB, 0, stream>>>(ws_y, zW, zB, out);
}
